// ReweightModel_34754875359655
// MI455X (gfx1250) — matrix operand order repaired and checked by reading, not compiled, not run on hardware
//
#include <hip/hip_runtime.h>
#include <hip/hip_bf16.h>
#include <stdint.h>

typedef __attribute__((ext_vector_type(16))) _Float16 v16h;
typedef __attribute__((ext_vector_type(8)))  float    v8f;

#define N_ROWS 100000
#define T_STEPS 60
#define DFEAT 5
#define HID 64
#define EMB 32
#define MCN 20
#define SIMS 32
#define LDIM 320          // D_FEAT*T + MC
#define GD 256            // 4*H
#define NE1 33            // E+1
#define PDIM 52           // E + MC
#define NPS 1562          // rows per sim
#define ROW0_SIM 50016    // N - n

union AFrag { v16h v; uint32_t u[8]; _Float16 h[16]; };

// gfx1250 has hardware V_TANH_F32 (TRANS op). Use it for tanh AND sigmoid
// (sigmoid(x) = 0.5*tanh(x/2)+0.5) to avoid the ~25-instruction softfloat
// exp/tanh expansion that dominated the VALU count.
__device__ __forceinline__ float fast_tanh(float x) {
#if __has_builtin(__builtin_amdgcn_tanhf)
  return __builtin_amdgcn_tanhf(x);
#else
  return tanhf(x);
#endif
}
__device__ __forceinline__ float fast_sigmoid(float x) {
  return 0.5f * fast_tanh(0.5f * x) + 0.5f;
}

__device__ __forceinline__ v8f wmma_f16(v16h a, v16h b, v8f c) {
  return __builtin_amdgcn_wmma_f32_16x16x32_f16(false, a, false, b, (short)0, c, false, false);
}

// B-matrix fragment (16-bit, 32x16 K x N): lane&15 = col, lanes>=16 hold K+16,
// VGPR v holds K = kbase + 2v (+16 for hi lanes). B[k][col] = W[col*ldW + k].
__device__ __forceinline__ v16h load_bfrag(const float* __restrict__ W, int ldW,
                                           int tile, int kbase, int kmax) {
  int lane = threadIdx.x & 31;
  int col = tile*16 + (lane & 15);
  int k0 = kbase + ((lane & 16) ? 16 : 0);
  v16h b;
  #pragma unroll
  for (int v = 0; v < 8; ++v) {
    int k = k0 + 2*v;
    float f0 = (k     < kmax) ? W[col*ldW + k]     : 0.f;
    float f1 = (k + 1 < kmax) ? W[col*ldW + k + 1] : 0.f;
    b[2*v]   = (_Float16)f0;
    b[2*v+1] = (_Float16)f1;
  }
  return b;
}

// A-matrix fragment (16-bit, 16x32 M x K) from f16 LDS buffer (row stride HID).
// lanes 0-15: M=lane, K 0..15; lanes 16-31: K 16..31.
__device__ __forceinline__ v16h load_afrag(const _Float16* __restrict__ buf) {
  int lane = threadIdx.x & 31;
  int m  = lane & 15;
  int kb = (lane & 16) ? 16 : 0;
  AFrag a;
  #pragma unroll
  for (int v = 0; v < 8; ++v) {
    int k = kb + 2*v;
    a.u[v] = *(const uint32_t*)(buf + m*HID + k);
  }
  return a.v;
}

// ---------------- K1: 2-layer LSTM (WMMA) + embedding + feature ----------------
__global__ __launch_bounds__(256)
void lstm_kernel(const float* __restrict__ data,
                 const float* __restrict__ Wih0, const float* __restrict__ Whh0,
                 const float* __restrict__ bih0, const float* __restrict__ bhh0,
                 const float* __restrict__ Wih1, const float* __restrict__ Whh1,
                 const float* __restrict__ bih1, const float* __restrict__ bhh1,
                 const float* __restrict__ linW, const float* __restrict__ linb,
                 const float* __restrict__ feW,  const float* __restrict__ feb,
                 float* __restrict__ emb_out, float* __restrict__ feat_out)
{
  __shared__ float    g_lds[16*GD];      // gate pre-activations 16x256
  __shared__ _Float16 h0_lds[16*HID];
  __shared__ _Float16 h1_lds[16*HID];
  __shared__ float    emb_lds[16*EMB];

  const int tid  = threadIdx.x;
  const int lane = tid & 31;
  const int wv   = tid >> 5;            // wave 0..7, owns column tiles 2wv,2wv+1
  const int row0 = blockIdx.x * 16;

  // Preload invariant weight B-fragments into registers (f16).
  v16h bih0_f[2], bhh0_f[2][2], bih1_f[2][2], bhh1_f[2][2];
  float bias0[2], bias1[2];
  #pragma unroll
  for (int t = 0; t < 2; ++t) {
    int ct = 2*wv + t;
    bih0_f[t] = load_bfrag(Wih0, DFEAT, ct, 0, DFEAT);   // K padded 5->32
    #pragma unroll
    for (int kh = 0; kh < 2; ++kh) {
      bhh0_f[t][kh] = load_bfrag(Whh0, HID, ct, 32*kh, HID);
      bih1_f[t][kh] = load_bfrag(Wih1, HID, ct, 32*kh, HID);
      bhh1_f[t][kh] = load_bfrag(Whh1, HID, ct, 32*kh, HID);
    }
    int col = ct*16 + (lane & 15);
    bias0[t] = bih0[col] + bhh0[col];
    bias1[t] = bih1[col] + bhh1[col];
  }

  for (int i = tid; i < 16*HID; i += 256) { h0_lds[i] = (_Float16)0.f; h1_lds[i] = (_Float16)0.f; }
  float c0[4], c1[4];
  #pragma unroll
  for (int j = 0; j < 4; ++j) { c0[j] = 0.f; c1[j] = 0.f; }
  __syncthreads();

  const float* xrow = data + (size_t)(row0 + (lane & 15)) * LDIM;

  for (int t = 0; t < T_STEPS; ++t) {
    // x_t A-fragment: 16 rows x K32 (5 valid features), zero padded
    AFrag ax;
    #pragma unroll
    for (int v = 0; v < 8; ++v) ax.u[v] = 0u;
    if (lane < 16) {
      #pragma unroll
      for (int f = 0; f < DFEAT; ++f) ax.h[f] = (_Float16)xrow[f*T_STEPS + t];
    }
    v16h ah0a = load_afrag(h0_lds);
    v16h ah0b = load_afrag(h0_lds + 32);

    // ---- layer 0 gates ----
    #pragma unroll
    for (int tt = 0; tt < 2; ++tt) {
      v8f acc;
      #pragma unroll
      for (int v = 0; v < 8; ++v) acc[v] = bias0[tt];
      acc = wmma_f16(ax.v, bih0_f[tt],    acc);
      acc = wmma_f16(ah0a, bhh0_f[tt][0], acc);
      acc = wmma_f16(ah0b, bhh0_f[tt][1], acc);
      int col = (2*wv + tt)*16 + (lane & 15);
      int rb  = (lane & 16) ? 8 : 0;
      #pragma unroll
      for (int v = 0; v < 8; ++v) g_lds[(rb + v)*GD + col] = acc[v];
    }
    __syncthreads();
    #pragma unroll
    for (int j = 0; j < 4; ++j) {
      int cell = tid + 256*j;
      int r = cell >> 6, hh = cell & 63;
      float gi = g_lds[r*GD + hh];
      float gf = g_lds[r*GD + 64  + hh];
      float gg = g_lds[r*GD + 128 + hh];
      float go = g_lds[r*GD + 192 + hh];
      float c = fast_sigmoid(gf)*c0[j] + fast_sigmoid(gi)*fast_tanh(gg);
      c0[j] = c;
      h0_lds[r*HID + hh] = (_Float16)(fast_sigmoid(go)*fast_tanh(c));
    }
    __syncthreads();

    // ---- layer 1 gates ----
    v16h axa  = load_afrag(h0_lds);
    v16h axb  = load_afrag(h0_lds + 32);
    v16h ah1a = load_afrag(h1_lds);
    v16h ah1b = load_afrag(h1_lds + 32);
    #pragma unroll
    for (int tt = 0; tt < 2; ++tt) {
      v8f acc;
      #pragma unroll
      for (int v = 0; v < 8; ++v) acc[v] = bias1[tt];
      acc = wmma_f16(axa,  bih1_f[tt][0], acc);
      acc = wmma_f16(axb,  bih1_f[tt][1], acc);
      acc = wmma_f16(ah1a, bhh1_f[tt][0], acc);
      acc = wmma_f16(ah1b, bhh1_f[tt][1], acc);
      int col = (2*wv + tt)*16 + (lane & 15);
      int rb  = (lane & 16) ? 8 : 0;
      #pragma unroll
      for (int v = 0; v < 8; ++v) g_lds[(rb + v)*GD + col] = acc[v];
    }
    __syncthreads();
    #pragma unroll
    for (int j = 0; j < 4; ++j) {
      int cell = tid + 256*j;
      int r = cell >> 6, hh = cell & 63;
      float gi = g_lds[r*GD + hh];
      float gf = g_lds[r*GD + 64  + hh];
      float gg = g_lds[r*GD + 128 + hh];
      float go = g_lds[r*GD + 192 + hh];
      float c = fast_sigmoid(gf)*c1[j] + fast_sigmoid(gi)*fast_tanh(gg);
      c1[j] = c;
      h1_lds[r*HID + hh] = (_Float16)(fast_sigmoid(go)*fast_tanh(c));
    }
    __syncthreads();
  }

  // embedding = tanh(h_last @ linW^T + linb), 16x32 outputs
  #pragma unroll
  for (int rr = 0; rr < 2; ++rr) {
    int o = tid + 256*rr;
    int r = o >> 5, e = o & 31;
    float acc = linb[e];
    #pragma unroll 8
    for (int k = 0; k < HID; ++k) acc += (float)h1_lds[r*HID + k] * linW[e*HID + k];
    float em = fast_tanh(acc);
    emb_lds[r*EMB + e] = em;
    emb_out[(size_t)(row0 + r)*EMB + e] = em;
  }
  __syncthreads();
  // feature = embedding @ feW^T + feb
  #pragma unroll
  for (int rr = 0; rr < 2; ++rr) {
    int o = tid + 256*rr;
    int r = o >> 5, e = o & 31;
    float acc = feb[e];
    #pragma unroll 8
    for (int k = 0; k < EMB; ++k) acc += emb_lds[r*EMB + k] * feW[e*EMB + k];
    feat_out[(size_t)(row0 + r)*EMB + e] = acc;
  }
}

// ---------------- K2: per-sim XtX (33x33) and XtY (33x20) ----------------
__global__ __launch_bounds__(256)
void xtx_kernel(const float* __restrict__ emb, const float* __restrict__ data,
                float* __restrict__ xtx, float* __restrict__ xty)
{
  __shared__ float xs[16][NE1 + 1];
  __shared__ float ys[16][MCN];
  const int s = blockIdx.x, tid = threadIdx.x;
  const int NENT = NE1*NE1 + NE1*MCN;   // 1749
  float acc[7];
  #pragma unroll
  for (int j = 0; j < 7; ++j) acc[j] = 0.f;

  for (int chunk = 0; chunk < (NPS + 15)/16; ++chunk) {
    __syncthreads();
    for (int idx = tid; idx < 16*(NE1 + MCN); idx += 256) {
      int r = idx / (NE1 + MCN), c = idx % (NE1 + MCN);
      int row = chunk*16 + r;
      float v = 0.f;
      if (row < NPS) {
        size_t grow = (size_t)(ROW0_SIM + s*NPS + row);
        if (c < NE1) v = (c == 0) ? 1.f : emb[grow*EMB + (c - 1)];
        else         v = data[grow*LDIM + 300 + (c - NE1)];
      }
      if (c < NE1) xs[r][c] = v; else ys[r][c - NE1] = v;
    }
    __syncthreads();
    #pragma unroll
    for (int j = 0; j < 7; ++j) {
      int ent = tid + 256*j;
      if (ent < NE1*NE1) {
        int i = ent / NE1, k = ent % NE1;
        float a = 0.f;
        #pragma unroll
        for (int r = 0; r < 16; ++r) a += xs[r][i]*xs[r][k];
        acc[j] += a;
      } else if (ent < NENT) {
        int e2 = ent - NE1*NE1;
        int i = e2 / MCN, m = e2 % MCN;
        float a = 0.f;
        #pragma unroll
        for (int r = 0; r < 16; ++r) a += xs[r][i]*ys[r][m];
        acc[j] += a;
      }
    }
  }
  #pragma unroll
  for (int j = 0; j < 7; ++j) {
    int ent = tid + 256*j;
    if (ent < NE1*NE1) xtx[s*NE1*NE1 + ent] = acc[j];
    else if (ent < NENT) xty[s*NE1*MCN + (ent - NE1*NE1)] = acc[j];
  }
}

// ---------------- K3: Gauss-Jordan solve xtx * ww = xty (per sim) ----------------
__global__ __launch_bounds__(64)
void solve_kernel(const float* __restrict__ xtx, const float* __restrict__ xty,
                  float* __restrict__ ww)
{
  __shared__ float A[NE1][NE1 + MCN + 1];
  __shared__ float fac[NE1];
  const int s = blockIdx.x, tid = threadIdx.x;
  const int NC = NE1 + MCN;   // 53
  for (int idx = tid; idx < NE1*NC; idx += 64) {
    int i = idx / NC, c = idx % NC;
    A[i][c] = (c < NE1) ? xtx[s*NE1*NE1 + i*NE1 + c]
                        : xty[s*NE1*MCN + i*MCN + (c - NE1)];
  }
  __syncthreads();
  for (int k = 0; k < NE1; ++k) {
    for (int i = tid; i < NE1; i += 64) fac[i] = A[i][k];
    __syncthreads();
    float inv = 1.f / fac[k];
    for (int c = tid; c < NC; c += 64) A[k][c] *= inv;
    __syncthreads();
    for (int idx = tid; idx < NE1*NC; idx += 64) {
      int i = idx / NC, c = idx % NC;
      if (i != k) A[i][c] -= fac[i]*A[k][c];
    }
    __syncthreads();
  }
  for (int idx = tid; idx < NE1*MCN; idx += 64) {
    int i = idx / MCN, m = idx % MCN;
    ww[s*NE1*MCN + i*MCN + m] = A[i][NE1 + m];
  }
}

// ---------------- K4: rwf over 32 sims, sigmoid weight, sum(w) ----------------
__global__ __launch_bounds__(256)
void rwf_kernel(const float* __restrict__ emb, const float* __restrict__ data,
                const float* __restrict__ ww,  const float* __restrict__ rwW,
                const float* __restrict__ rwb,
                float* __restrict__ wout, float* __restrict__ S0)
{
  __shared__ float wl[8*NE1*MCN];        // 8-sim chunk of ww, 21 KB
  __shared__ float errs[256][MCN + 1];   // per-thread error row
  __shared__ float red[256];
  const int tid = threadIdx.x;
  const int row = blockIdx.x*256 + tid;
  const bool valid = row < N_ROWS;

  float e1[NE1];
  e1[0] = 1.f;
  #pragma unroll
  for (int i = 0; i < EMB; ++i) e1[i + 1] = valid ? emb[(size_t)row*EMB + i] : 0.f;
  #pragma unroll
  for (int m = 0; m < MCN; ++m) errs[tid][m] = valid ? data[(size_t)row*LDIM + 300 + m] : 0.f;

  float rwf[SIMS];
  #pragma unroll
  for (int sc = 0; sc < 4; ++sc) {
    __syncthreads();
    for (int idx = tid; idx < 8*NE1*MCN; idx += 256) wl[idx] = ww[sc*8*NE1*MCN + idx];
    __syncthreads();
    #pragma unroll
    for (int s0 = 0; s0 < 8; ++s0) {
      float accsum = 0.f;
      for (int m = 0; m < MCN; ++m) {
        float p = 0.f;
        #pragma unroll
        for (int i = 0; i < NE1; ++i) p += e1[i]*wl[s0*NE1*MCN + i*MCN + m];
        float d = p - errs[tid][m];
        accsum += d*d;
      }
      rwf[sc*8 + s0] = 0.5f * accsum * (1.0f/(float)MCN);
    }
  }
  float pre = rwb[0];
  #pragma unroll
  for (int s = 0; s < SIMS; ++s) pre += rwf[s]*rwW[s];
  float wv = valid ? fast_sigmoid(pre) : 0.f;
  if (valid) wout[row] = wv;
  red[tid] = wv; __syncthreads();
  for (int off = 128; off > 0; off >>= 1) {
    if (tid < off) red[tid] += red[tid + off];
    __syncthreads();
  }
  if (tid == 0) atomicAdd(S0, red[0]);
}

// ---------------- K5: weighted moments S1 = sum w*p, S2 = sum w*p*p^T ----------------
__global__ __launch_bounds__(256)
void moment_kernel(const float* __restrict__ feat, const float* __restrict__ data,
                   const float* __restrict__ wv,
                   float* __restrict__ S1, float* __restrict__ S2)
{
  __shared__ float pw[16][PDIM + 1];
  __shared__ float wls[16];
  const int tid = threadIdx.x;
  const int rbase = blockIdx.x * 1024;
  const int NSLOT = PDIM + PDIM*PDIM;   // 2756
  float acc[11];
  #pragma unroll
  for (int j = 0; j < 11; ++j) acc[j] = 0.f;

  for (int chunk = 0; chunk < 64; ++chunk) {
    __syncthreads();
    int r0 = rbase + chunk*16;
    for (int idx = tid; idx < 16*(PDIM + 1); idx += 256) {
      int r = idx / (PDIM + 1), c = idx % (PDIM + 1);
      int row = r0 + r;
      float v = 0.f;
      if (row < N_ROWS) {
        if (c < EMB)       v = feat[(size_t)row*EMB + c];
        else if (c < PDIM) v = data[(size_t)row*LDIM + 300 + (c - EMB)];
        else               v = wv[row];
      }
      if (c < PDIM) pw[r][c] = v; else wls[r] = v;
    }
    __syncthreads();
    #pragma unroll
    for (int j = 0; j < 11; ++j) {
      int slot = tid + 256*j;
      if (slot < NSLOT) {
        float a = 0.f;
        if (slot < PDIM) {
          #pragma unroll
          for (int r = 0; r < 16; ++r) a += wls[r]*pw[r][slot];
        } else {
          int e2 = slot - PDIM; int jj = e2 / PDIM, kk = e2 % PDIM;
          #pragma unroll
          for (int r = 0; r < 16; ++r) a += wls[r]*pw[r][jj]*pw[r][kk];
        }
        acc[j] += a;
      }
    }
  }
  #pragma unroll
  for (int j = 0; j < 11; ++j) {
    int slot = tid + 256*j;
    if (slot < PDIM) atomicAdd(&S1[slot], acc[j]);
    else if (slot < NSLOT) atomicAdd(&S2[slot - PDIM], acc[j]);
  }
}

// ---------------- K6: finalize miu / sig ----------------
__global__ __launch_bounds__(256)
void final_kernel(const float* __restrict__ S0, const float* __restrict__ S1,
                  const float* __restrict__ S2, float* __restrict__ out)
{
  __shared__ float miu[PDIM];
  const int tid = threadIdx.x;
  float invs = 1.f / S0[0];
  if (tid < PDIM) { miu[tid] = S1[tid]*invs; out[tid] = miu[tid]; }
  __syncthreads();
  for (int idx = tid; idx < PDIM*PDIM; idx += 256) {
    int j = idx / PDIM, k = idx % PDIM;
    out[PDIM + idx] = S2[idx]*invs - miu[j]*miu[k];
  }
}

__global__ void zero_kernel(float* p, int n) {
  int i = blockIdx.x*256 + threadIdx.x;
  if (i < n) p[i] = 0.f;
}

extern "C" void kernel_launch(void* const* d_in, const int* in_sizes, int n_in,
                              void* d_out, int out_size, void* d_ws, size_t ws_size,
                              hipStream_t stream)
{
  const float* data = (const float*)d_in[0];
  const float* Wih0 = (const float*)d_in[1];
  const float* Whh0 = (const float*)d_in[2];
  const float* bih0 = (const float*)d_in[3];
  const float* bhh0 = (const float*)d_in[4];
  const float* Wih1 = (const float*)d_in[5];
  const float* Whh1 = (const float*)d_in[6];
  const float* bih1 = (const float*)d_in[7];
  const float* bhh1 = (const float*)d_in[8];
  const float* linW = (const float*)d_in[9];
  const float* linb = (const float*)d_in[10];
  const float* feW  = (const float*)d_in[11];
  const float* feb  = (const float*)d_in[12];
  const float* rwW  = (const float*)d_in[13];
  const float* rwb  = (const float*)d_in[14];

  float* ws   = (float*)d_ws;
  float* emb  = ws;
  float* feat = emb  + (size_t)N_ROWS*EMB;
  float* wv   = feat + (size_t)N_ROWS*EMB;
  float* xtx  = wv   + N_ROWS;
  float* xty  = xtx  + SIMS*NE1*NE1;
  float* wwp  = xty  + SIMS*NE1*MCN;
  float* S0   = wwp  + SIMS*NE1*MCN;
  float* S1   = S0 + 1;
  float* S2   = S1 + PDIM;

  lstm_kernel<<<N_ROWS/16, 256, 0, stream>>>(data, Wih0, Whh0, bih0, bhh0,
                                             Wih1, Whh1, bih1, bhh1,
                                             linW, linb, feW, feb, emb, feat);
  zero_kernel<<<(1 + PDIM + PDIM*PDIM + 255)/256, 256, 0, stream>>>(S0, 1 + PDIM + PDIM*PDIM);
  xtx_kernel<<<SIMS, 256, 0, stream>>>(emb, data, xtx, xty);
  solve_kernel<<<SIMS, 64, 0, stream>>>(xtx, xty, wwp);
  rwf_kernel<<<(N_ROWS + 255)/256, 256, 0, stream>>>(emb, data, wwp, rwW, rwb, wv, S0);
  moment_kernel<<<(N_ROWS + 1023)/1024, 256, 0, stream>>>(feat, data, wv, S1, S2);
  final_kernel<<<1, 256, 0, stream>>>(S0, S1, S2, (float*)d_out);
}
